// TailFeedForward_9929964389245
// MI455X (gfx1250) — matrix operand order repaired and checked by reading, not compiled, not run on hardware
//
#include <hip/hip_runtime.h>
#include <math.h>

#define NTOK 4096
#define HD   768
#define FD   3072
#define NE   4
#define CAP  1024

#define BM 128
#define BN 128
#define BK 32

typedef __attribute__((ext_vector_type(16))) __bf16 v16bf;
typedef __attribute__((ext_vector_type(8)))  float  v8f;

union Frag {
  uint4 u[2];
  v16bf v;
};

__device__ __forceinline__ unsigned short f2bf(float f) {
  unsigned int u = __float_as_uint(f);
  u += 0x7FFFu + ((u >> 16) & 1u);   // round-to-nearest-even
  return (unsigned short)(u >> 16);
}

// ---------------------------------------------------------------------------
// Kernel 1: routers (common + unique). One block per token.
// ---------------------------------------------------------------------------
__global__ void __launch_bounds__(256)
router_kernel(const float* __restrict__ h,
              const float* __restrict__ Wsc, const float* __restrict__ bsc,
              const float* __restrict__ Wsu, const float* __restrict__ bsu,
              int* __restrict__ route_c, float* __restrict__ pmax_c,
              int* __restrict__ route_u, float* __restrict__ pmax_u) {
  __shared__ float red[256][8];
  const int i   = blockIdx.x;
  const int tid = threadIdx.x;
  const float* hr = h + (size_t)i * HD;

  float acc[8] = {0.f,0.f,0.f,0.f,0.f,0.f,0.f,0.f};
  for (int k = tid; k < HD; k += 256) {
    float hv = hr[k];
#pragma unroll
    for (int e = 0; e < NE; ++e) {
      acc[e]      += hv * Wsc[k * NE + e];
      acc[NE + e] += hv * Wsu[k * NE + e];
    }
  }
#pragma unroll
  for (int e = 0; e < 8; ++e) red[tid][e] = acc[e];
  __syncthreads();
  for (int s = 128; s > 0; s >>= 1) {
    if (tid < s) {
#pragma unroll
      for (int e = 0; e < 8; ++e) red[tid][e] += red[tid + s][e];
    }
    __syncthreads();
  }
  if (tid == 0) {
    float lc[NE], lu[NE];
#pragma unroll
    for (int e = 0; e < NE; ++e) { lc[e] = red[0][e] + bsc[e]; lu[e] = red[0][NE + e] + bsu[e]; }
    int ac = 0; float mc = lc[0];
#pragma unroll
    for (int e = 1; e < NE; ++e) if (lc[e] > mc) { mc = lc[e]; ac = e; }
    float sc = 0.f;
#pragma unroll
    for (int e = 0; e < NE; ++e) sc += expf(lc[e] - mc);
    route_c[i] = ac; pmax_c[i] = 1.0f / sc;

    int au = 0; float mu = lu[0];
#pragma unroll
    for (int e = 1; e < NE; ++e) if (lu[e] > mu) { mu = lu[e]; au = e; }
    float su = 0.f;
#pragma unroll
    for (int e = 0; e < NE; ++e) su += expf(lu[e] - mu);
    route_u[i] = au; pmax_u[i] = 1.0f / su;
  }
}

// ---------------------------------------------------------------------------
// Kernel 2: capacity ranking (exact lexsort semantics). One block per token.
// ---------------------------------------------------------------------------
__global__ void __launch_bounds__(256)
rank_kernel(const int* __restrict__ route_c, const float* __restrict__ pmax_c,
            int* __restrict__ dropped) {
  __shared__ int sred[256];
  const int i   = blockIdx.x;
  const int tid = threadIdx.x;
  const int   ri = route_c[i];
  const float pi = pmax_c[i];
  int cnt = 0;
  for (int j = tid; j < NTOK; j += 256) {
    if (route_c[j] == ri) {
      float pj = pmax_c[j];
      if (pj > pi || (pj == pi && j < i)) cnt++;
    }
  }
  sred[tid] = cnt;
  __syncthreads();
  for (int s = 128; s > 0; s >>= 1) {
    if (tid < s) sred[tid] += sred[tid + s];
    __syncthreads();
  }
  if (tid == 0) dropped[i] = (sred[0] >= CAP) ? 1 : 0;
}

// ---------------------------------------------------------------------------
// Kernel 3: build per-expert token lists (order-free, atomic compaction).
// ---------------------------------------------------------------------------
__global__ void __launch_bounds__(256)
compact_kernel(const int* __restrict__ route_c, const int* __restrict__ route_u,
               const int* __restrict__ dropped,
               int* __restrict__ cnt_c, int* __restrict__ cnt_u,
               int* __restrict__ list_c, int* __restrict__ list_u) {
  int i = blockIdx.x * blockDim.x + threadIdx.x;
  if (i >= NTOK) return;
  int ec  = route_c[i];
  int pos = atomicAdd(&cnt_c[ec], 1);
  list_c[ec * NTOK + pos] = i;
  if (dropped[i]) {
    int eu = route_u[i];
    int pu = atomicAdd(&cnt_u[eu], 1);
    list_u[eu * NTOK + pu] = i;
  }
}

// ---------------------------------------------------------------------------
// Kernel 4: grouped GEMM1: Y[tok, 0:3072] = gelu(h[tok] @ W1[e] + b1[e])
// grid = (FD/BN, 32, NE); 256 threads (8 waves); bf16 WMMA, fp32 accumulate.
// ---------------------------------------------------------------------------
__global__ void __launch_bounds__(256)
gemm1_kernel(const float* __restrict__ h, const float* __restrict__ W1,
             const float* __restrict__ b1, const int* __restrict__ cnt,
             const int* __restrict__ list, unsigned short* __restrict__ Y) {
  const int e  = blockIdx.z;
  const int ne = cnt[e];
  const int rt = blockIdx.y;
  if (rt * BM >= ne) return;
  const int n0  = blockIdx.x * BN;
  const int tid = threadIdx.x;

  __shared__ int toks[BM];
  __shared__ __align__(16) unsigned short As[BM][BK];   // [row][k]  8 KB
  __shared__ __align__(16) unsigned short Bs[BN][BK];   // [col][k]  8 KB (transposed)

  if (tid < BM) {
    int r = rt * BM + tid;
    toks[tid] = (r < ne) ? list[e * NTOK + r] : -1;
  }
  __syncthreads();

  const float* We  = W1 + (size_t)e * HD * FD;
  const float* b1e = b1 + e * FD;

  const int lane = tid & 31;
  const int wid  = tid >> 5;
  const int wr = wid & 3, wc = wid >> 2;      // 4x2 wave grid -> 32x64 per wave
  const int lh = lane & 15, sel = lane >> 4;

  v8f acc[2][4];
#pragma unroll
  for (int i = 0; i < 2; ++i)
#pragma unroll
    for (int j = 0; j < 4; ++j) {
      v8f z = {0.f,0.f,0.f,0.f,0.f,0.f,0.f,0.f};
      acc[i][j] = z;
    }

  for (int k0 = 0; k0 < HD; k0 += BK) {
    // --- stage A tile (gathered token rows, fp32 -> bf16) ---
    for (int idx = tid; idx < (BM * BK) / 4; idx += 256) {
      int r = idx >> 3;
      int q = idx & 7;
      int tk = toks[r];
      float4 v = make_float4(0.f, 0.f, 0.f, 0.f);
      if (tk >= 0) v = *(const float4*)(h + (size_t)tk * HD + k0 + q * 4);
      unsigned int lo = (unsigned)f2bf(v.x) | ((unsigned)f2bf(v.y) << 16);
      unsigned int hi = (unsigned)f2bf(v.z) | ((unsigned)f2bf(v.w) << 16);
      *(uint2*)&As[r][q * 4] = make_uint2(lo, hi);
    }
    // --- stage B tile transposed (W1 row-major K x N, fp32 -> bf16) ---
    for (int idx = tid; idx < (BK * BN) / 4; idx += 256) {
      int k = idx >> 5;
      int q = idx & 31;
      float4 v = *(const float4*)(We + (size_t)(k0 + k) * FD + n0 + q * 4);
      int n = q * 4;
      Bs[n + 0][k] = f2bf(v.x);
      Bs[n + 1][k] = f2bf(v.y);
      Bs[n + 2][k] = f2bf(v.z);
      Bs[n + 3][k] = f2bf(v.w);
    }
    __syncthreads();

    Frag a[2], b[4];
#pragma unroll
    for (int i = 0; i < 2; ++i) {
      const unsigned short* ap = &As[wr * 32 + i * 16 + lh][sel * 8];
      a[i].u[0] = *(const uint4*)ap;          // K = sel*8   .. sel*8+7
      a[i].u[1] = *(const uint4*)(ap + 16);   // K = 16+sel*8 .. +7
    }
#pragma unroll
    for (int j = 0; j < 4; ++j) {
      const unsigned short* bp = &Bs[wc * 64 + j * 16 + lh][sel * 8];
      b[j].u[0] = *(const uint4*)bp;          // K = sel*8   .. sel*8+7
      b[j].u[1] = *(const uint4*)(bp + 16);   // K = 16+sel*8 .. +7
    }
#pragma unroll
    for (int i = 0; i < 2; ++i)
#pragma unroll
      for (int j = 0; j < 4; ++j)
        acc[i][j] = __builtin_amdgcn_wmma_f32_16x16x32_bf16(
            false, a[i].v, false, b[j].v, (short)0, acc[i][j], false, false);
    __syncthreads();
  }

  // --- bias + exact GELU, store bf16 Y (indexed by original token id) ---
#pragma unroll
  for (int i = 0; i < 2; ++i) {
#pragma unroll
    for (int v = 0; v < 8; ++v) {
      int rloc = wr * 32 + i * 16 + sel * 8 + v;
      int tk = toks[rloc];
      if (tk < 0) continue;
      unsigned short* yrow = Y + (size_t)tk * FD;
#pragma unroll
      for (int j = 0; j < 4; ++j) {
        int col = n0 + wc * 64 + j * 16 + lh;
        float x = acc[i][j][v] + b1e[col];
        float g = 0.5f * x * (1.0f + erff(x * 0.70710678118654752440f));
        yrow[col] = f2bf(g);
      }
    }
  }
}

// ---------------------------------------------------------------------------
// Kernel 5: grouped GEMM2: out[tok, 0:768] (=|+=) Y[tok] @ W2[e] + b2[e]
// grid = (HD/BN, 32, NE)
// ---------------------------------------------------------------------------
__global__ void __launch_bounds__(256)
gemm2_kernel(const unsigned short* __restrict__ Y, const float* __restrict__ W2,
             const float* __restrict__ b2, const int* __restrict__ cnt,
             const int* __restrict__ list, float* __restrict__ out, int accum) {
  const int e  = blockIdx.z;
  const int ne = cnt[e];
  const int rt = blockIdx.y;
  if (rt * BM >= ne) return;
  const int n0  = blockIdx.x * BN;
  const int tid = threadIdx.x;

  __shared__ int toks[BM];
  __shared__ __align__(16) unsigned short As[BM][BK];
  __shared__ __align__(16) unsigned short Bs[BN][BK];

  if (tid < BM) {
    int r = rt * BM + tid;
    toks[tid] = (r < ne) ? list[e * NTOK + r] : -1;
  }
  __syncthreads();

  const float* We  = W2 + (size_t)e * FD * HD;
  const float* b2e = b2 + e * HD;

  const int lane = tid & 31;
  const int wid  = tid >> 5;
  const int wr = wid & 3, wc = wid >> 2;
  const int lh = lane & 15, sel = lane >> 4;

  v8f acc[2][4];
#pragma unroll
  for (int i = 0; i < 2; ++i)
#pragma unroll
    for (int j = 0; j < 4; ++j) {
      v8f z = {0.f,0.f,0.f,0.f,0.f,0.f,0.f,0.f};
      acc[i][j] = z;
    }

  for (int k0 = 0; k0 < FD; k0 += BK) {
    // --- stage A tile (gathered bf16 rows of Y) ---
    for (int idx = tid; idx < (BM * BK) / 4; idx += 256) {
      int r = idx >> 3;
      int q = idx & 7;
      int tk = toks[r];
      uint2 v = make_uint2(0u, 0u);
      if (tk >= 0) v = *(const uint2*)(Y + (size_t)tk * FD + k0 + q * 4);
      *(uint2*)&As[r][q * 4] = v;
    }
    // --- stage B tile transposed (W2 row-major FD x HD, fp32 -> bf16) ---
    for (int idx = tid; idx < (BK * BN) / 4; idx += 256) {
      int k = idx >> 5;
      int q = idx & 31;
      float4 v = *(const float4*)(We + (size_t)(k0 + k) * HD + n0 + q * 4);
      int n = q * 4;
      Bs[n + 0][k] = f2bf(v.x);
      Bs[n + 1][k] = f2bf(v.y);
      Bs[n + 2][k] = f2bf(v.z);
      Bs[n + 3][k] = f2bf(v.w);
    }
    __syncthreads();

    Frag a[2], b[4];
#pragma unroll
    for (int i = 0; i < 2; ++i) {
      const unsigned short* ap = &As[wr * 32 + i * 16 + lh][sel * 8];
      a[i].u[0] = *(const uint4*)ap;
      a[i].u[1] = *(const uint4*)(ap + 16);
    }
#pragma unroll
    for (int j = 0; j < 4; ++j) {
      const unsigned short* bp = &Bs[wc * 64 + j * 16 + lh][sel * 8];
      b[j].u[0] = *(const uint4*)bp;
      b[j].u[1] = *(const uint4*)(bp + 16);
    }
#pragma unroll
    for (int i = 0; i < 2; ++i)
#pragma unroll
      for (int j = 0; j < 4; ++j)
        acc[i][j] = __builtin_amdgcn_wmma_f32_16x16x32_bf16(
            false, a[i].v, false, b[j].v, (short)0, acc[i][j], false, false);
    __syncthreads();
  }

#pragma unroll
  for (int i = 0; i < 2; ++i) {
#pragma unroll
    for (int v = 0; v < 8; ++v) {
      int rloc = wr * 32 + i * 16 + sel * 8 + v;
      int tk = toks[rloc];
      if (tk < 0) continue;
      float* orow = out + (size_t)tk * HD;
#pragma unroll
      for (int j = 0; j < 4; ++j) {
        int col = n0 + wc * 64 + j * 16 + lh;
        float x = acc[i][j][v] + b2e[col];
        if (accum) orow[col] += x;
        else       orow[col] = x;
      }
    }
  }
}

// ---------------------------------------------------------------------------
// Host launcher
// ---------------------------------------------------------------------------
extern "C" void kernel_launch(void* const* d_in, const int* in_sizes, int n_in,
                              void* d_out, int out_size, void* d_ws, size_t ws_size,
                              hipStream_t stream) {
  const float* h   = (const float*)d_in[0];
  const float* Wsc = (const float*)d_in[1];
  const float* bsc = (const float*)d_in[2];
  const float* W1c = (const float*)d_in[3];
  const float* b1c = (const float*)d_in[4];
  const float* W2c = (const float*)d_in[5];
  const float* b2c = (const float*)d_in[6];
  const float* Wsu = (const float*)d_in[7];
  const float* bsu = (const float*)d_in[8];
  const float* W1u = (const float*)d_in[9];
  const float* b1u = (const float*)d_in[10];
  const float* W2u = (const float*)d_in[11];
  const float* b2u = (const float*)d_in[12];
  float* out = (float*)d_out;

  // workspace layout
  char* w = (char*)d_ws;
  int*   route_c = (int*)(w + 0 * (1 << 14));
  int*   route_u = (int*)(w + 1 * (1 << 14));
  int*   dropped = (int*)(w + 2 * (1 << 14));
  float* pmax_c  = (float*)(w + 3 * (1 << 14));
  float* pmax_u  = (float*)(w + 4 * (1 << 14));
  char*  cntbase = w + 5 * (1 << 14);            // 128 bytes of counters
  int*   cnt_c   = (int*)(cntbase);
  int*   cnt_u   = (int*)(cntbase + 64);
  int*   list_c  = (int*)(cntbase + 128);
  int*   list_u  = (int*)(cntbase + 128 + (NE * NTOK) * sizeof(int));
  unsigned short* Y =
      (unsigned short*)(cntbase + 128 + 2 * (NE * NTOK) * sizeof(int));   // 24 MB bf16

  // 1) routers
  router_kernel<<<NTOK, 256, 0, stream>>>(h, Wsc, bsc, Wsu, bsu,
                                          route_c, pmax_c, route_u, pmax_u);
  // 2) capacity ranking -> dropped mask
  rank_kernel<<<NTOK, 256, 0, stream>>>(route_c, pmax_c, dropped);
  // 3) compaction
  hipMemsetAsync(cntbase, 0, 128, stream);
  compact_kernel<<<NTOK / 256, 256, 0, stream>>>(route_c, route_u, dropped,
                                                 cnt_c, cnt_u, list_c, list_u);
  // 4) common FFN (all tokens): writes out
  gemm1_kernel<<<dim3(FD / BN, NTOK / BM, NE), 256, 0, stream>>>(h, W1c, b1c,
                                                                 cnt_c, list_c, Y);
  gemm2_kernel<<<dim3(HD / BN, NTOK / BM, NE), 256, 0, stream>>>(Y, W2c, b2c,
                                                                 cnt_c, list_c, out, 0);
  // 5) unique FFN (dropped tokens only): accumulates into out (reuses Y; stream-ordered)
  gemm1_kernel<<<dim3(FD / BN, NTOK / BM, NE), 256, 0, stream>>>(h, W1u, b1u,
                                                                 cnt_u, list_u, Y);
  gemm2_kernel<<<dim3(HD / BN, NTOK / BM, NE), 256, 0, stream>>>(Y, W2u, b2u,
                                                                 cnt_u, list_u, out, 1);
}
